// PerceiverIO_MoE_36679020708519
// MI455X (gfx1250) — compile-verified, matrix operand order checked by reading, not run on hardware
//
#include <hip/hip_runtime.h>
#include <hip/hip_bf16.h>
#include <math.h>

typedef __attribute__((ext_vector_type(16))) _Float16 v16h;
typedef __attribute__((ext_vector_type(8)))  _Float16 v8h;
typedef __attribute__((ext_vector_type(8)))  float    v8f;

#define NB   2048
#define DD   64
#define NE   4
#define NR   8
#define NH   4
#define NL   4
#define NCC  7

#define KPMAX 320   // max padded K (text: 300 -> 320)
#define KSTR  328   // LDS row stride in halfs (656B = 164 dwords; bank-friendly)

// ---------------------------------------------------------------------------
// Kernel A: mean over time.  src [B,T,F] -> dst [B,F].
// This is the roofline term: 543 MB read ~= 23 us at 23.3 TB/s.
// Templated on vector width so each lane issues global_load_b128/b64:
//   text  F=300: VEC=4 (row stride 1200B, 16B-aligned everywhere)
//   audio F=74 : VEC=2 (stride 296B, 8B-aligned)
//   video F=35 : VEC=1
// One thread per (b, vec-column); consecutive threads -> consecutive
// addresses each t (coalesced); 4 vector accumulators keep 4 loads in flight.
// ---------------------------------------------------------------------------
template <int VEC>
__global__ __launch_bounds__(256)
void pool_mean_kernel(const float* __restrict__ src, float* __restrict__ dst,
                      int Bn, int T, int F) {
  typedef float vf __attribute__((ext_vector_type(VEC)));
  const int nc = F / VEC;                     // F is a multiple of VEC by choice
  int idx = blockIdx.x * blockDim.x + threadIdx.x;
  if (idx >= Bn * nc) return;
  int b = idx / nc, c = idx - b * nc;
  const float* base = src + (size_t)b * T * F + (size_t)c * VEC;
  vf s0 = {}, s1 = {}, s2 = {}, s3 = {};
  int t = 0;
  for (; t + 4 <= T; t += 4) {
    if (t + 16 < T) __builtin_prefetch(base + (size_t)(t + 16) * F, 0, 0);
    s0 += *(const vf*)(base + (size_t)(t + 0) * F);
    s1 += *(const vf*)(base + (size_t)(t + 1) * F);
    s2 += *(const vf*)(base + (size_t)(t + 2) * F);
    s3 += *(const vf*)(base + (size_t)(t + 3) * F);
  }
  for (; t < T; ++t) s0 += *(const vf*)(base + (size_t)t * F);
  vf s = (s0 + s1) + (s2 + s3);
  s *= (1.0f / (float)T);
  *(vf*)(dst + (size_t)b * F + (size_t)c * VEC) = s;
}

// ---------------------------------------------------------------------------
// Kernel B: x[B,F] @ w[F,64] + bias, then LayerNorm + ReLU, written into
// inputs[B,3,64] slot.  Block = 128 threads = 4 waves; block covers 16 batch
// rows x 64 cols; wave w owns cols 16w..16w+15.
//
// Staging: As[16][K] f16 activation tile, Bs[64][K] f16 = W^T, both padded
// with zeros to Kp (multiple of 32).  Inner loop is branch-free: per K-step
// each lane does 4x ds_load_b128 (16B contiguous half-runs) + 1x v_wmma.
//
// Fragment layouts (16-bit, 16x32), ISA 7.12.2:
//   A: lane<16 -> M=lane, K kq..kq+7 & kq+16..kq+23 with kq=0; lane>=16 -> kq=8.
//   B: symmetric striping in N (N = lane&15, same kq split).
// C layout: VGPR i -> M = i + 8*(lane>=16), N = lane&15.
// ---------------------------------------------------------------------------
__global__ __launch_bounds__(128)
void proj_ln_relu_wmma(const float* __restrict__ x, const float* __restrict__ w,
                       const float* __restrict__ bias,
                       const float* __restrict__ g, const float* __restrict__ bb,
                       float* __restrict__ dst, int F, int slot) {
  __shared__ __align__(16) _Float16 As[16][KSTR];
  __shared__ __align__(16) _Float16 Bs[DD][KSTR];
  __shared__ float Ot[16][DD];

  const int tid  = threadIdx.x;
  const int wave = tid >> 5;
  const int lane = tid & 31;
  const int b0   = blockIdx.x * 16;
  const int m    = lane & 15;
  const int kq   = (lane >> 4) * 8;
  const int ncol = wave * 16 + (lane & 15);
  const int Kp   = ((F + 31) >> 5) << 5;

  // cooperative fill (coalesced global reads; zero-pad K in [F, Kp))
  for (int i = tid; i < 16 * Kp; i += 128) {
    int mm = i / Kp, k = i - mm * Kp;
    As[mm][k] = (k < F) ? (_Float16)x[(size_t)(b0 + mm) * F + k] : (_Float16)0.f;
  }
  for (int i = tid; i < DD * Kp; i += 128) {
    int k = i >> 6, n = i & 63;                 // n fastest -> coalesced w read
    Bs[n][k] = (k < F) ? (_Float16)w[(size_t)k * DD + n] : (_Float16)0.f;
  }
  __syncthreads();

  v8f c = {};
  for (int kb = 0; kb < Kp; kb += 32) {
    v8h a0  = *(const v8h*)&As[m][kb + kq];
    v8h a1  = *(const v8h*)&As[m][kb + kq + 16];
    v8h bm0 = *(const v8h*)&Bs[ncol][kb + kq];
    v8h bm1 = *(const v8h*)&Bs[ncol][kb + kq + 16];
    v16h a  = __builtin_shufflevector(a0, a1, 0, 1, 2, 3, 4, 5, 6, 7,
                                      8, 9, 10, 11, 12, 13, 14, 15);
    v16h bm = __builtin_shufflevector(bm0, bm1, 0, 1, 2, 3, 4, 5, 6, 7,
                                      8, 9, 10, 11, 12, 13, 14, 15);
    c = __builtin_amdgcn_wmma_f32_16x16x32_f16(false, a, false, bm,
                                               (short)0, c, false, false);
  }
  __syncthreads();

#pragma unroll
  for (int i = 0; i < 8; ++i) {
    int mr = i + ((lane >> 4) << 3);
    Ot[mr][ncol] = c[i] + bias[ncol];
  }
  __syncthreads();

  if (tid < 16) {
    float mu = 0.f;
    for (int d = 0; d < DD; ++d) mu += Ot[tid][d];
    mu *= (1.0f / DD);
    float va = 0.f;
    for (int d = 0; d < DD; ++d) { float df = Ot[tid][d] - mu; va += df * df; }
    va *= (1.0f / DD);
    float inv = rsqrtf(va + 1e-5f);
    float* o = dst + ((size_t)(b0 + tid) * 3 + slot) * DD;
    for (int d = 0; d < DD; ++d) {
      float y = (Ot[tid][d] - mu) * inv * g[d] + bb[d];
      o[d] = fmaxf(y, 0.f);
    }
  }
}

// ---------------------------------------------------------------------------
// Kernel C: everything after `inputs` for one batch row per block.
// ---------------------------------------------------------------------------
__global__ __launch_bounds__(128)
void fused_head_kernel(
    const float* __restrict__ inputs, const float* __restrict__ latent,
    const float* __restrict__ wq, const float* __restrict__ bq,
    const float* __restrict__ wk, const float* __restrict__ bk,
    const float* __restrict__ wv, const float* __restrict__ bv,
    const float* __restrict__ wo, const float* __restrict__ bo,
    const float* __restrict__ n1g, const float* __restrict__ n1b,
    const float* __restrict__ fw1, const float* __restrict__ fb1,
    const float* __restrict__ fw2, const float* __restrict__ fb2,
    const float* __restrict__ n2g, const float* __restrict__ n2b,
    const float* __restrict__ rw1, const float* __restrict__ rb1,
    const float* __restrict__ rlg, const float* __restrict__ rlb,
    const float* __restrict__ rw2, const float* __restrict__ rb2,
    const float* __restrict__ p1w, const float* __restrict__ p1b,
    const float* __restrict__ p2w, const float* __restrict__ p2b,
    const float* __restrict__ ld, const float* __restrict__ lu,
    const float* __restrict__ clg, const float* __restrict__ clb,
    const float* __restrict__ cw1, const float* __restrict__ cb1,
    const float* __restrict__ cw2, const float* __restrict__ cb2,
    float* __restrict__ out_logits, float* __restrict__ out_w,
    float* __restrict__ out_fused,  float* __restrict__ out_f,
    float* __restrict__ out_v1,     float* __restrict__ out_v2) {
  const int b = blockIdx.x;
  const int tid = threadIdx.x;

  __shared__ float inp[3][DD], qs[NL][DD], ks[3][DD], vs[3][DD];
  __shared__ float att[NH][NL][3];
  __shared__ float ao[NL][DD], lat[NL][DD], hn[NL][4 * DD], lat2[NL][DD];
  __shared__ float fv[DD], t1[DD], rlog[NE], wts[NE], dn[NE][NR];
  __shared__ float fus[DD], cn[DD], hh[DD];
  __shared__ float sc2[2];

  for (int i = tid; i < 3 * DD; i += 128) inp[i / DD][i % DD] = inputs[(size_t)b * 3 * DD + i];
  // q = latent @ wq + bq  (batch-invariant; tiny, recomputed per block)
  for (int i = tid; i < NL * DD; i += 128) {
    int l = i >> 6, d = i & 63;
    float s = bq[d];
    for (int k2 = 0; k2 < DD; ++k2) s += latent[l * DD + k2] * wq[k2 * DD + d];
    qs[l][d] = s;
  }
  __syncthreads();
  for (int i = tid; i < 3 * DD; i += 128) {
    int s3 = i >> 6, d = i & 63;
    float sk = bk[d], sv = bv[d];
    for (int k2 = 0; k2 < DD; ++k2) {
      float xv = inp[s3][k2];
      sk += xv * wk[k2 * DD + d];
      sv += xv * wv[k2 * DD + d];
    }
    ks[s3][d] = sk; vs[s3][d] = sv;
  }
  __syncthreads();
  if (tid < NH * NL) {
    int h = tid >> 2, l = tid & 3;
    float sc[3], mx = -1e30f;
    for (int s3 = 0; s3 < 3; ++s3) {
      float s = 0.f;
      for (int d2 = 0; d2 < 16; ++d2) s += qs[l][h * 16 + d2] * ks[s3][h * 16 + d2];
      sc[s3] = s * 0.25f;
      mx = fmaxf(mx, sc[s3]);
    }
    float den = 0.f;
    for (int s3 = 0; s3 < 3; ++s3) { sc[s3] = expf(sc[s3] - mx); den += sc[s3]; }
    for (int s3 = 0; s3 < 3; ++s3) att[h][l][s3] = sc[s3] / den;
  }
  __syncthreads();
  for (int i = tid; i < NL * DD; i += 128) {
    int l = i >> 6, d = i & 63, h = d >> 4;
    float s = 0.f;
    for (int s3 = 0; s3 < 3; ++s3) s += att[h][l][s3] * vs[s3][d];
    ao[l][d] = s;
  }
  __syncthreads();
  for (int i = tid; i < NL * DD; i += 128) {
    int l = i >> 6, d = i & 63;
    float s = bo[d];
    for (int c2 = 0; c2 < DD; ++c2) s += ao[l][c2] * wo[c2 * DD + d];
    lat[l][d] = latent[l * DD + d] + s;
  }
  __syncthreads();
  if (tid < NL) {
    float mu = 0.f; for (int d = 0; d < DD; ++d) mu += lat[tid][d]; mu *= (1.0f / DD);
    float va = 0.f; for (int d = 0; d < DD; ++d) { float df = lat[tid][d] - mu; va += df * df; }
    float inv = rsqrtf(va * (1.0f / DD) + 1e-5f);
    for (int d = 0; d < DD; ++d) lat[tid][d] = (lat[tid][d] - mu) * inv * n1g[d] + n1b[d];
  }
  __syncthreads();
  for (int i = tid; i < NL * 4 * DD; i += 128) {
    int l = i >> 8, j = i & 255;
    float s = fb1[j];
    for (int c2 = 0; c2 < DD; ++c2) s += lat[l][c2] * fw1[c2 * 4 * DD + j];
    hn[l][j] = 0.5f * s * (1.0f + erff(s * 0.70710678118f));
  }
  __syncthreads();
  for (int i = tid; i < NL * DD; i += 128) {
    int l = i >> 6, d = i & 63;
    float s = fb2[d];
    for (int j = 0; j < 4 * DD; ++j) s += hn[l][j] * fw2[j * DD + d];
    lat2[l][d] = lat[l][d] + s;
  }
  __syncthreads();
  if (tid < NL) {
    float mu = 0.f; for (int d = 0; d < DD; ++d) mu += lat2[tid][d]; mu *= (1.0f / DD);
    float va = 0.f; for (int d = 0; d < DD; ++d) { float df = lat2[tid][d] - mu; va += df * df; }
    float inv = rsqrtf(va * (1.0f / DD) + 1e-5f);
    for (int d = 0; d < DD; ++d) lat2[tid][d] = (lat2[tid][d] - mu) * inv * n2g[d] + n2b[d];
  }
  __syncthreads();
  if (tid < DD) fv[tid] = 0.25f * (lat2[0][tid] + lat2[1][tid] + lat2[2][tid] + lat2[3][tid]);
  __syncthreads();
  if (tid < DD) {
    float s = rb1[tid];
    for (int c2 = 0; c2 < DD; ++c2) s += fv[c2] * rw1[c2 * DD + tid];
    t1[tid] = s;
  }
  __syncthreads();
  if (tid == 0) {
    float mu = 0.f; for (int d = 0; d < DD; ++d) mu += t1[d]; mu *= (1.0f / DD);
    float va = 0.f; for (int d = 0; d < DD; ++d) { float df = t1[d] - mu; va += df * df; }
    sc2[0] = mu; sc2[1] = rsqrtf(va * (1.0f / DD) + 1e-5f);
  }
  __syncthreads();
  if (tid < DD) t1[tid] = fmaxf((t1[tid] - sc2[0]) * sc2[1] * rlg[tid] + rlb[tid], 0.f);
  __syncthreads();
  if (tid < NE) {
    float s = rb2[tid];
    for (int d2 = 0; d2 < DD; ++d2) s += t1[d2] * rw2[d2 * NE + tid];
    rlog[tid] = s;
  }
  __syncthreads();
  if (tid == 0) {
    float mx = fmaxf(fmaxf(rlog[0], rlog[1]), fmaxf(rlog[2], rlog[3]));
    float den = 0.f, e2[NE];
    for (int e = 0; e < NE; ++e) { e2[e] = expf(rlog[e] - mx); den += e2[e]; }
    for (int e = 0; e < NE; ++e) wts[e] = e2[e] / den;
  }
  if (tid < DD / 2) {
    float s1 = p1b[tid], s2 = p2b[tid];
    for (int d2 = 0; d2 < DD; ++d2) {
      s1 += fv[d2] * p1w[d2 * (DD / 2) + tid];
      s2 += fv[d2] * p2w[d2 * (DD / 2) + tid];
    }
    out_v1[(size_t)b * (DD / 2) + tid] = s1;
    out_v2[(size_t)b * (DD / 2) + tid] = s2;
  }
  if (tid >= 64 && tid < 96) {
    int t = tid - 64, e = t >> 3, r = t & 7;
    float s = 0.f;
    for (int d2 = 0; d2 < DD; ++d2) s += fv[d2] * ld[(e * DD + d2) * NR + r];
    dn[e][r] = s;
  }
  __syncthreads();
  if (tid < DD) {
    float acc = 0.f;
    for (int e = 0; e < NE; ++e) {
      float up = 0.f;
      for (int r = 0; r < NR; ++r) up += dn[e][r] * lu[(e * NR + r) * DD + tid];
      acc += wts[e] * (fv[tid] + 2.0f * up);   // SCALING = 16/8 = 2
    }
    fus[tid] = acc;
    out_fused[(size_t)b * DD + tid] = acc;
    out_f[(size_t)b * DD + tid] = fv[tid];
  }
  if (tid < NE) out_w[(size_t)b * NE + tid] = wts[tid];
  __syncthreads();
  if (tid == 0) {
    float mu = 0.f; for (int d = 0; d < DD; ++d) mu += fus[d]; mu *= (1.0f / DD);
    float va = 0.f; for (int d = 0; d < DD; ++d) { float df = fus[d] - mu; va += df * df; }
    sc2[0] = mu; sc2[1] = rsqrtf(va * (1.0f / DD) + 1e-5f);
  }
  __syncthreads();
  if (tid < DD) cn[tid] = (fus[tid] - sc2[0]) * sc2[1] * clg[tid] + clb[tid];
  __syncthreads();
  if (tid < DD) {
    float s = cb1[tid];
    for (int d2 = 0; d2 < DD; ++d2) s += cn[d2] * cw1[d2 * DD + tid];
    hh[tid] = fmaxf(s, 0.f);
  }
  __syncthreads();
  if (tid < NCC) {
    float s = cb2[tid];
    for (int d2 = 0; d2 < DD; ++d2) s += hh[d2] * cw2[d2 * NCC + tid];
    out_logits[(size_t)b * NCC + tid] = s;
  }
}

// ---------------------------------------------------------------------------
extern "C" void kernel_launch(void* const* d_in, const int* in_sizes, int n_in,
                              void* d_out, int out_size, void* d_ws, size_t ws_size,
                              hipStream_t stream) {
  const float* text  = (const float*)d_in[0];
  const float* audio = (const float*)d_in[1];
  const float* video = (const float*)d_in[2];
  const float* text_w = (const float*)d_in[3],  *text_b = (const float*)d_in[4];
  const float* text_lg = (const float*)d_in[5], *text_lb = (const float*)d_in[6];
  const float* aud_w = (const float*)d_in[7],   *aud_b = (const float*)d_in[8];
  const float* aud_lg = (const float*)d_in[9],  *aud_lb = (const float*)d_in[10];
  const float* vid_w = (const float*)d_in[11],  *vid_b = (const float*)d_in[12];
  const float* vid_lg = (const float*)d_in[13], *vid_lb = (const float*)d_in[14];
  const float* latent = (const float*)d_in[15];
  const float* wq = (const float*)d_in[16], *bq = (const float*)d_in[17];
  const float* wk = (const float*)d_in[18], *bk = (const float*)d_in[19];
  const float* wv = (const float*)d_in[20], *bv = (const float*)d_in[21];
  const float* wo = (const float*)d_in[22], *bo = (const float*)d_in[23];
  const float* n1g = (const float*)d_in[24], *n1b = (const float*)d_in[25];
  const float* fw1 = (const float*)d_in[26], *fb1 = (const float*)d_in[27];
  const float* fw2 = (const float*)d_in[28], *fb2 = (const float*)d_in[29];
  const float* n2g = (const float*)d_in[30], *n2b = (const float*)d_in[31];
  const float* rw1 = (const float*)d_in[32], *rb1 = (const float*)d_in[33];
  const float* rlg = (const float*)d_in[34], *rlb = (const float*)d_in[35];
  const float* rw2 = (const float*)d_in[36], *rb2 = (const float*)d_in[37];
  const float* p1w = (const float*)d_in[38], *p1b = (const float*)d_in[39];
  const float* p2w = (const float*)d_in[40], *p2b = (const float*)d_in[41];
  const float* ld  = (const float*)d_in[42], *lu  = (const float*)d_in[43];
  const float* clg = (const float*)d_in[44], *clb = (const float*)d_in[45];
  const float* cw1 = (const float*)d_in[46], *cb1 = (const float*)d_in[47];
  const float* cw2 = (const float*)d_in[48], *cb2 = (const float*)d_in[49];

  float* ws = (float*)d_ws;
  float* pt = ws;                               // [2048,300]
  float* pa = pt + (size_t)NB * 300;            // [2048,74]
  float* pv = pa + (size_t)NB * 74;             // [2048,35]
  float* inputs = pv + (size_t)NB * 35;         // [2048,3,64]

  // 1) bandwidth-bound mean pooling (543 MB read; this IS the kernel)
  {
    int nct = 300 / 4, nca = 74 / 2, ncv = 35;
    pool_mean_kernel<4><<<(NB * nct + 255) / 256, 256, 0, stream>>>(text,  pt, NB, 128, 300);
    pool_mean_kernel<2><<<(NB * nca + 255) / 256, 256, 0, stream>>>(audio, pa, NB, 256, 74);
    pool_mean_kernel<1><<<(NB * ncv + 255) / 256, 256, 0, stream>>>(video, pv, NB, 256, 35);
  }

  // 2) WMMA projections + LN + ReLU -> inputs[B,3,64]
  proj_ln_relu_wmma<<<NB / 16, 128, 0, stream>>>(pt, text_w, text_b, text_lg, text_lb, inputs, 300, 0);
  proj_ln_relu_wmma<<<NB / 16, 128, 0, stream>>>(pa, aud_w,  aud_b,  aud_lg,  aud_lb,  inputs, 74,  1);
  proj_ln_relu_wmma<<<NB / 16, 128, 0, stream>>>(pv, vid_w,  vid_b,  vid_lg,  vid_lb,  inputs, 35,  2);

  // 3) fused perceiver / router / LoRA / classifier, one block per batch row
  float* out = (float*)d_out;
  float* out_logits = out;                       // [2048,7]
  float* out_w      = out + (size_t)NB * NCC;    // [2048,4]
  float* out_fused  = out_w + (size_t)NB * NE;   // [2048,64]
  float* out_f      = out_fused + (size_t)NB * DD;
  float* out_v1     = out_f + (size_t)NB * DD;   // [2048,32]
  float* out_v2     = out_v1 + (size_t)NB * (DD / 2);
  fused_head_kernel<<<NB, 128, 0, stream>>>(
      inputs, latent, wq, bq, wk, bk, wv, bv, wo, bo, n1g, n1b,
      fw1, fb1, fw2, fb2, n2g, n2b, rw1, rb1, rlg, rlb, rw2, rb2,
      p1w, p1b, p2w, p2b, ld, lu, clg, clb, cw1, cb1, cw2, cb2,
      out_logits, out_w, out_fused, out_f, out_v1, out_v2);
}
